// self_deep_matching_10531259810264
// MI455X (gfx1250) — matrix operand order repaired and checked by reading, not compiled, not run on hardware
//
#include <hip/hip_runtime.h>

// ---------------------------------------------------------------------------
// CDNA5 (gfx1250) implementation of the SCSA deep-matching network.
// All contractions (conv-as-im2col-GEMM + attention GEMMs) run on
// v_wmma_f32_16x16x32_bf16 (bf16 A/B, f32 accumulate).
// ---------------------------------------------------------------------------

typedef __attribute__((ext_vector_type(16))) __bf16 v16bf;
typedef __attribute__((ext_vector_type(8)))  float  v8f;

union Frag16 { unsigned short u[16]; unsigned int d[8]; v16bf v; };

__device__ __forceinline__ unsigned short f2bf(float f) {
  unsigned u = __float_as_uint(f);
  u += 0x7FFFu + ((u >> 16) & 1u);          // round-to-nearest-even
  return (unsigned short)(u >> 16);
}

// ---------------------------------------------------------------------------
// Generic bf16 GEMM:  C[M,N] = A[M,K] * B[K,N]   (row-major, strided)
// Block = 256 threads = 8 waves. Block tile 64x128. Wave tile 32x32
// (4 WMMA accumulators, fragments reused). K stepped by 32.
//
// LDS:
//   lA: 64 rows x 32 k   stored as 20 dwords/row (16 data + 4 pad, 16B-aligned
//       rows) -> each A fragment is two ds_load_b128.
//   lB: 16 k-pairs x 128 cols, dword = B[2k][n] | B[2k+1][n]<<16 (+4 pad)
//       -> each B fragment element is one ds_load_b32.
// ---------------------------------------------------------------------------
__global__ __launch_bounds__(256)
void k_gemm_bf16(const unsigned short* __restrict__ A, int lda,
                 const unsigned short* __restrict__ B, int ldb,
                 float* __restrict__ C, int ldc,
                 int M, int N, int K) {
  __shared__ unsigned int lA[64 * 20];
  __shared__ unsigned int lB[16 * 132];

  const int tid  = threadIdx.x;
  const int lane = tid & 31;
  const int wv   = tid >> 5;
  const int mBase = blockIdx.y * 64;
  const int nBase = blockIdx.x * 128;
  const int mOff  = (wv & 1) * 32;        // wave M offset within tile
  const int nOff  = (wv >> 1) * 32;       // wave N offset within tile
  const int half  = lane >> 4;            // 0: lanes 0-15, 1: lanes 16-31
  const int l15   = lane & 15;

  v8f acc00 = {}, acc01 = {}, acc10 = {}, acc11 = {};

  const bool vecA = (mBase + 64 <= M) && ((lda & 7) == 0);
  const bool vecB = (nBase + 128 <= N) && ((ldb & 3) == 0);

  for (int k0 = 0; k0 < K; k0 += 32) {
    const bool fullK = (k0 + 32 <= K);

    // ---------------- A tile (64 x 32) ----------------
    if (vecA && fullK) {
      int r = tid >> 2, j = tid & 3;      // one uint4 (8 bf16) per thread
      const uint4* src = (const uint4*)(A + (long)(mBase + r) * lda + k0) + j;
      *(uint4*)&lA[r * 20 + j * 4] = *src;
    } else {
      for (int e = tid; e < 64 * 32; e += 256) {
        int r = e >> 5, c = e & 31;
        int gr = mBase + r, gc = k0 + c;
        unsigned short v =
            (gr < M && gc < K) ? A[(long)gr * lda + gc] : (unsigned short)0;
        ((unsigned short*)&lA[r * 20])[c] = v;
      }
    }

    // ---------------- B tile (32 x 128, K-paired) ----------------
    if (vecB && fullK) {
      for (int u = tid; u < 16 * 32; u += 256) {   // (kpair, 4-col group)
        int kp = u >> 5, cq = (u & 31) * 4;
        const unsigned short* b0 = B + (long)(k0 + 2 * kp) * ldb + nBase + cq;
        const unsigned short* b1 = b0 + ldb;
        uint2 lo = *(const uint2*)b0;
        uint2 hi = *(const uint2*)b1;
        unsigned int o0 = (lo.x & 0xffffu) | (hi.x << 16);
        unsigned int o1 = (lo.x >> 16)     | (hi.x & 0xffff0000u);
        unsigned int o2 = (lo.y & 0xffffu) | (hi.y << 16);
        unsigned int o3 = (lo.y >> 16)     | (hi.y & 0xffff0000u);
        *(uint4*)&lB[kp * 132 + cq] = make_uint4(o0, o1, o2, o3);
      }
    } else {
      for (int e = tid; e < 16 * 128; e += 256) {
        int kp = e >> 7, c = e & 127;
        int gr0 = k0 + 2 * kp, gr1 = gr0 + 1, gc = nBase + c;
        unsigned int lo = (gr0 < K && gc < N) ? B[(long)gr0 * ldb + gc] : 0u;
        unsigned int hi = (gr1 < K && gc < N) ? B[(long)gr1 * ldb + gc] : 0u;
        lB[kp * 132 + c] = lo | (hi << 16);
      }
    }

    // speculative prefetch of the next K slab
    if (k0 + 32 < K) {
      int pr = mBase + (tid & 63);
      if (pr < M) __builtin_prefetch(A + (long)pr * lda + k0 + 32, 0, 1);
      int pk = k0 + 32 + (tid & 31);
      if (pk < K) __builtin_prefetch(B + (long)pk * ldb + nBase, 0, 1);
    }
    __syncthreads();

    // ---------------- fragments (CDNA5 16-bit WMMA layouts) ----------------
    Frag16 fa0, fa1, fb0, fb1;
    {
      const unsigned int* pa0 = &lA[(mOff + l15) * 20 + 4 * half];
      const unsigned int* pa1 = &lA[(mOff + 16 + l15) * 20 + 4 * half];
#pragma unroll
      for (int v = 0; v < 4; ++v) {       // K = 8*half + 2v (+1)
        fa0.d[v] = pa0[v];
        fa1.d[v] = pa1[v];
      }
#pragma unroll
      for (int v = 0; v < 4; ++v) {       // K = 16 + 8*half + 2v (+1)
        fa0.d[4 + v] = pa0[8 + v];
        fa1.d[4 + v] = pa1[8 + v];
      }
#pragma unroll
      for (int v = 0; v < 8; ++v) {       // kpair = 4*half + v (+4 for v >= 4)
        int kp = 4 * half + v + (v >= 4 ? 4 : 0);
        fb0.d[v] = lB[kp * 132 + nOff + l15];
        fb1.d[v] = lB[kp * 132 + nOff + 16 + l15];
      }
    }
    acc00 = __builtin_amdgcn_wmma_f32_16x16x32_bf16(false, fa0.v, false, fb0.v,
                                                    (short)0, acc00, false, false);
    acc01 = __builtin_amdgcn_wmma_f32_16x16x32_bf16(false, fa0.v, false, fb1.v,
                                                    (short)0, acc01, false, false);
    acc10 = __builtin_amdgcn_wmma_f32_16x16x32_bf16(false, fa1.v, false, fb0.v,
                                                    (short)0, acc10, false, false);
    acc11 = __builtin_amdgcn_wmma_f32_16x16x32_bf16(false, fa1.v, false, fb1.v,
                                                    (short)0, acc11, false, false);
    __syncthreads();
  }

  // C/D layout: VGPR r -> row M = r + 8*half, col N = lane&15
  const int gcol0 = nBase + nOff + l15;
  const int gcol1 = gcol0 + 16;
#pragma unroll
  for (int r = 0; r < 8; ++r) {
    int grow0 = mBase + mOff + r + 8 * half;
    int grow1 = grow0 + 16;
    if (grow0 < M) {
      if (gcol0 < N) C[(long)grow0 * ldc + gcol0] = acc00[r];
      if (gcol1 < N) C[(long)grow0 * ldc + gcol1] = acc01[r];
    }
    if (grow1 < M) {
      if (gcol0 < N) C[(long)grow1 * ldc + gcol0] = acc10[r];
      if (gcol1 < N) C[(long)grow1 * ldc + gcol1] = acc11[r];
    }
  }
}

// ---------------------------------------------------------------------------
// Elementwise / data-movement kernels
// ---------------------------------------------------------------------------
__global__ void k_convert_bf16(const float* __restrict__ s,
                               unsigned short* __restrict__ d, long n) {
  long n2 = n >> 1;
  for (long i = blockIdx.x * (long)blockDim.x + threadIdx.x; i < n2;
       i += (long)gridDim.x * blockDim.x) {
    float2 f = ((const float2*)s)[i];
    unsigned int lo = f2bf(f.x), hi = f2bf(f.y);
    ((unsigned int*)d)[i] = lo | (hi << 16);
  }
  if ((n & 1) && blockIdx.x == 0 && threadIdx.x == 0) d[n - 1] = f2bf(s[n - 1]);
}

// src (R x Cc) f32 row-major  ->  dst (Cc x R) bf16 row-major
__global__ void k_transpose_bf16(const float* __restrict__ src,
                                 unsigned short* __restrict__ dst,
                                 int R, int Cc, long total) {
  for (long i = blockIdx.x * (long)blockDim.x + threadIdx.x; i < total;
       i += (long)gridDim.x * blockDim.x) {
    int r = (int)(i / Cc), c = (int)(i % Cc);
    dst[(long)c * R + r] = f2bf(src[i]);
  }
}

// per-channel spatial (H,W) transpose: dst[t][x*H+y] = src[t][y*W+x]
__global__ void k_spatial_transpose_bf16(const float* __restrict__ src,
                                         unsigned short* __restrict__ dst,
                                         int H, int W, long total) {
  long hw = (long)H * W;
  for (long i = blockIdx.x * (long)blockDim.x + threadIdx.x; i < total;
       i += (long)gridDim.x * blockDim.x) {
    long t = i / hw; long rem = i % hw;
    int y = (int)(rem / W), x = (int)(rem % W);
    dst[t * hw + (long)x * H + y] = f2bf(src[i]);
  }
}

__global__ void k_bias_act(float* __restrict__ x, const float* __restrict__ b,
                           long HW, int relu, long total) {
  for (long i = blockIdx.x * (long)blockDim.x + threadIdx.x; i < total;
       i += (long)gridDim.x * blockDim.x) {
    int c = (int)(i / HW);
    float v = x[i] + b[c];
    if (relu) v = fmaxf(v, 0.f);
    x[i] = v;
  }
}

__global__ void k_maxpool2(const float* __restrict__ in, float* __restrict__ out,
                           int H, int W, long total) {
  int Ho = H >> 1, Wo = W >> 1;
  long hwo = (long)Ho * Wo;
  for (long i = blockIdx.x * (long)blockDim.x + threadIdx.x; i < total;
       i += (long)gridDim.x * blockDim.x) {
    long c = i / hwo; long rem = i % hwo;
    int y = (int)(rem / Wo), x = (int)(rem % Wo);
    const float* p = in + (c * H + 2 * y) * (long)W + 2 * x;
    out[i] = fmaxf(fmaxf(p[0], p[1]), fmaxf(p[W], p[W + 1]));
  }
}

__global__ void k_upsample2(const float* __restrict__ in, float* __restrict__ out,
                            int H, int W, long total) {
  int Ho = H * 2, Wo = W * 2;
  long hwo = (long)Ho * Wo;
  for (long i = blockIdx.x * (long)blockDim.x + threadIdx.x; i < total;
       i += (long)gridDim.x * blockDim.x) {
    long c = i / hwo; long rem = i % hwo;
    int y = (int)(rem / Wo), x = (int)(rem % Wo);
    out[i] = in[(c * H + (y >> 1)) * (long)W + (x >> 1)];
  }
}

// im2col, f32 activation -> bf16 column matrix  col[K x chunk], K = Cin*ks*ks
__global__ void k_im2col_bf16(const float* __restrict__ in,
                              unsigned short* __restrict__ col,
                              int H, int W, int ks, int pad, int dil,
                              long n0, int chunk, long total) {
  int kk = ks * ks;
  for (long i = blockIdx.x * (long)blockDim.x + threadIdx.x; i < total;
       i += (long)gridDim.x * blockDim.x) {
    int j = (int)(i % chunk);
    int k = (int)(i / chunk);
    long n = n0 + j;
    int y = (int)(n / W), x = (int)(n % W);
    int c = k / kk, rem = k % kk;
    int kh = rem / ks, kw = rem % ks;
    int iy = y - pad + kh * dil, ix = x - pad + kw * dil;
    float v = 0.f;
    if (iy >= 0 && iy < H && ix >= 0 && ix < W)
      v = in[((long)c * H + iy) * W + ix];
    col[(long)k * chunk + j] = f2bf(v);
  }
}

// fn = x / sqrt(sum_c x^2) per spatial position; emits f32 + bf16 copies
__global__ void k_frob_spatial(const float* __restrict__ x,
                               float* __restrict__ fn,
                               unsigned short* __restrict__ fnb,
                               int C, int HW) {
  __shared__ float red[256];
  int n = blockIdx.x, t = threadIdx.x;
  float s = 0.f;
  for (int c = t; c < C; c += 256) { float v = x[(long)c * HW + n]; s += v * v; }
  red[t] = s; __syncthreads();
  for (int o = 128; o > 0; o >>= 1) { if (t < o) red[t] += red[t + o]; __syncthreads(); }
  float inv = rsqrtf(red[0]);
  for (int c = t; c < C; c += 256) {
    float v = x[(long)c * HW + n] * inv;
    fn[(long)c * HW + n] = v;
    fnb[(long)c * HW + n] = f2bf(v);
  }
}

__global__ void k_softmax_rows(float* __restrict__ s, int N) {
  __shared__ float red[256];
  long row = blockIdx.x; int t = threadIdx.x;
  float* p = s + row * (long)N;
  float m = -3.4e38f;
  for (int i = t; i < N; i += 256) m = fmaxf(m, p[i]);
  red[t] = m; __syncthreads();
  for (int o = 128; o > 0; o >>= 1) { if (t < o) red[t] = fmaxf(red[t], red[t + o]); __syncthreads(); }
  m = red[0]; __syncthreads();
  float sum = 0.f;
  for (int i = t; i < N; i += 256) { float e = __expf(p[i] - m); p[i] = e; sum += e; }
  red[t] = sum; __syncthreads();
  for (int o = 128; o > 0; o >>= 1) { if (t < o) red[t] += red[t + o]; __syncthreads(); }
  float inv = 1.0f / red[0];
  for (int i = t; i < N; i += 256) p[i] *= inv;
}

// fdd[n*C+c] = scale * o[n*C+c] + fn[c*HW+n]   (+ bf16 copy)
__global__ void k_scale_add(const float* __restrict__ o,
                            const float* __restrict__ fn,
                            const float* __restrict__ scale,
                            float* __restrict__ fdd,
                            unsigned short* __restrict__ fddb,
                            int C, int HW, long total) {
  float sc = scale[0];
  for (long i = blockIdx.x * (long)blockDim.x + threadIdx.x; i < total;
       i += (long)gridDim.x * blockDim.x) {
    long n = i / C; int c = (int)(i % C);
    float v = sc * o[i] + fn[(long)c * HW + n];
    fdd[i] = v; fddb[i] = f2bf(v);
  }
}

// iterative top-48 (descending) + relu; one block per row of cm (N<=4096)
__global__ void k_topk48(const float* __restrict__ cm, float* __restrict__ top, int N) {
  __shared__ float row[4096];
  __shared__ float rv[256];
  __shared__ int   ri[256];
  long r = blockIdx.x; int t = threadIdx.x;
  for (int i = t; i < N; i += 256) row[i] = cm[r * (long)N + i];
  __syncthreads();
  for (int it = 0; it < 48; ++it) {
    float bv = -3.4e38f; int bi = 0;
    for (int i = t; i < N; i += 256) { float v = row[i]; if (v > bv) { bv = v; bi = i; } }
    rv[t] = bv; ri[t] = bi; __syncthreads();
    for (int o = 128; o > 0; o >>= 1) {
      if (t < o && rv[t + o] > rv[t]) { rv[t] = rv[t + o]; ri[t] = ri[t + o]; }
      __syncthreads();
    }
    if (t == 0) { top[r * 48 + it] = fmaxf(rv[0], 0.f); row[ri[0]] = -3.4e38f; }
    __syncthreads();
  }
}

// denom[w*48+t] = sum_h top[(h*W+w)*48+t]^2
__global__ void k_top_norm(const float* __restrict__ top, float* __restrict__ denom,
                           int H, int W) {
  int i = blockIdx.x * blockDim.x + threadIdx.x;
  if (i >= W * 48) return;
  int w = i / 48, t = i % 48;
  float s = 0.f;
  for (int h = 0; h < H; ++h) { float v = top[((long)(h * W + w)) * 48 + t]; s += v * v; }
  denom[i] = s;
}

// out[t*HW + n] = top[n*48+t] / sqrt(denom[w*48+t])   (transpose BHWT -> BTHW)
__global__ void k_top_write(const float* __restrict__ top,
                            const float* __restrict__ denom,
                            float* __restrict__ out, int H, int W, long total) {
  long hw = (long)H * W;
  for (long i = blockIdx.x * (long)blockDim.x + threadIdx.x; i < total;
       i += (long)gridDim.x * blockDim.x) {
    int t = (int)(i % 48); long n = i / 48; int w = (int)(n % W);
    out[(long)t * hw + n] = top[n * 48 + t] * rsqrtf(denom[w * 48 + t]);
  }
}

__global__ void k_channel_mean(const float* __restrict__ x, float* __restrict__ avg, int HW) {
  __shared__ float red[256];
  int c = blockIdx.x, t = threadIdx.x;
  float s = 0.f;
  for (int n = t; n < HW; n += 256) s += x[(long)c * HW + n];
  red[t] = s; __syncthreads();
  for (int o = 128; o > 0; o >>= 1) { if (t < o) red[t] += red[t + o]; __syncthreads(); }
  if (t == 0) avg[c] = red[0] / HW;
}

// ASSP pooled branch: 3x3 conv on a padded 1x1 map == center-tap matvec; bcast
__global__ void k_assp_pool(const float* __restrict__ avg, const float* __restrict__ w,
                            const float* __restrict__ b, float* __restrict__ out,
                            int Cin, int HW) {
  int o = blockIdx.x;
  float v = 0.f;
  for (int i = 0; i < Cin; ++i) v += w[((long)o * Cin + i) * 9 + 4] * avg[i];
  v = fmaxf(v + b[o], 0.f);
  for (int n = threadIdx.x; n < HW; n += blockDim.x) out[(long)o * HW + n] = v;
}

// ---------------------------------------------------------------------------
// Host-side helpers (all launches on `stream`; no allocation, no sync)
// ---------------------------------------------------------------------------
static inline int nblk(long n) {
  long b = (n + 255) / 256;
  if (b < 1) b = 1;
  if (b > 262144) b = 262144;
  return (int)b;
}

static void gemm(const unsigned short* A, int lda, const unsigned short* B, int ldb,
                 float* C, int ldc, int M, int N, int K, hipStream_t st) {
  dim3 g((N + 127) / 128, (M + 63) / 64), b(256);
  k_gemm_bf16<<<g, b, 0, st>>>(A, lda, B, ldb, C, ldc, M, N, K);
}

static void cvt(const float* s, unsigned short* d, long n, hipStream_t st) {
  k_convert_bf16<<<nblk(n / 2 + 1), 256, 0, st>>>(s, d, n);
}

static void biasact(float* x, const float* b, int C, long HW, bool relu, hipStream_t st) {
  long total = (long)C * HW;
  k_bias_act<<<nblk(total), 256, 0, st>>>(x, b, HW, relu ? 1 : 0, total);
}

static const long COLCAP = 48L * 1024 * 1024;  // bf16 elements in col scratch

// Generic conv layer: out[Cout, H, W] = conv(in[Cin, H, W]); 'same' spatial size.
static void conv_layer(const float* in, int Cin, int H, int W,
                       const float* w, const float* b, int Cout,
                       int ks, int pad, int dil, bool relu, float* out,
                       unsigned short* wb, unsigned short* colb, hipStream_t st) {
  int Kd = Cin * ks * ks;
  long N = (long)H * W;
  cvt(w, wb, (long)Cout * Kd, st);
  if (ks == 1 && pad == 0) {
    cvt(in, colb, (long)Cin * N, st);
    gemm(wb, Kd, colb, (int)N, out, (int)N, Cout, (int)N, Kd, st);
  } else {
    long chunkMax = COLCAP / Kd;
    chunkMax &= ~127L;
    if (chunkMax < 128) chunkMax = 128;
    if (chunkMax > N) chunkMax = N;
    for (long n0 = 0; n0 < N; n0 += chunkMax) {
      int chunk = (int)((N - n0 < chunkMax) ? (N - n0) : chunkMax);
      long total = (long)Kd * chunk;
      k_im2col_bf16<<<nblk(total), 256, 0, st>>>(in, colb, H, W, ks, pad, dil,
                                                 n0, chunk, total);
      gemm(wb, Kd, colb, chunk, out + n0, (int)N, Cout, chunk, Kd, st);
    }
  }
  biasact(out, b, Cout, N, relu, st);
}

struct Scratch {
  unsigned short *wb, *colb;
  float* sbuf;            // 4096x4096 f32 (s, then reused for c)
  unsigned short* betab;  // 4096x4096 bf16
  float* fnf; unsigned short* fnb;
  float *fo, *go, *ho;
  unsigned short *fT, *gT, *hT;
  float *obuf, *fdd; unsigned short* fddb;
  float *topb, *denom;
};

// SCSA block on a (C, 64, 64) feature map; writes 48 channels at outc.
static void scsa(const float* x, int C, void* const* din, int base,
                 float* outc, const Scratch& S, hipStream_t st) {
  const float* f_b   = (const float*)din[base + 0];
  const float* f_w   = (const float*)din[base + 1];
  const float* g_b   = (const float*)din[base + 2];
  const float* g_w   = (const float*)din[base + 3];
  const float* h_b   = (const float*)din[base + 4];
  const float* h_w   = (const float*)din[base + 5];
  const float* scale = (const float*)din[base + 6];
  const int H = 64, Wd = 64, HW = 4096;
  const int tc = C / 8;

  k_frob_spatial<<<HW, 256, 0, st>>>(x, S.fnf, S.fnb, C, HW);

  // 1x1 convs f,g,h  (Wf[tc,C] @ fn[C,HW])
  cvt(f_w, S.wb, (long)tc * C, st);
  gemm(S.wb, C, S.fnb, HW, S.fo, HW, tc, HW, C, st);
  biasact(S.fo, f_b, tc, HW, false, st);
  cvt(g_w, S.wb, (long)tc * C, st);
  gemm(S.wb, C, S.fnb, HW, S.go, HW, tc, HW, C, st);
  biasact(S.go, g_b, tc, HW, false, st);
  cvt(h_w, S.wb, (long)C * C, st);
  gemm(S.wb, C, S.fnb, HW, S.ho, HW, C, HW, C, st);
  biasact(S.ho, h_b, C, HW, false, st);

  // fT = f^T (HW x tc); gT = spatially-transposed g (tc x HW); hT = h^T (HW x C)
  k_transpose_bf16<<<nblk((long)tc * HW), 256, 0, st>>>(S.fo, S.fT, tc, HW, (long)tc * HW);
  k_spatial_transpose_bf16<<<nblk((long)tc * HW), 256, 0, st>>>(S.go, S.gT, H, Wd, (long)tc * HW);
  k_transpose_bf16<<<nblk((long)C * HW), 256, 0, st>>>(S.ho, S.hT, C, HW, (long)C * HW);

  // s = fT @ gT   (HW x HW, K = tc)
  gemm(S.fT, tc, S.gT, HW, S.sbuf, HW, HW, HW, tc, st);
  k_softmax_rows<<<HW, 256, 0, st>>>(S.sbuf, HW);
  cvt(S.sbuf, S.betab, (long)HW * HW, st);

  // o = beta @ hT   (HW x C, K = HW)
  gemm(S.betab, HW, S.hT, C, S.obuf, C, HW, C, HW, st);

  // fdd = scale*o + fn^T   (HW x C layout, raw buffer also viewed as C x HW)
  k_scale_add<<<nblk((long)HW * C), 256, 0, st>>>(S.obuf, S.fnf, scale,
                                                  S.fdd, S.fddb, C, HW, (long)HW * C);

  // c = fdd (HW x C) @ fdd-as-(C x HW)  -> reuse sbuf
  gemm(S.fddb, C, S.fddb, HW, S.sbuf, HW, HW, HW, C, st);

  k_topk48<<<HW, 256, 0, st>>>(S.sbuf, S.topb, HW);
  k_top_norm<<<(Wd * 48 + 255) / 256, 256, 0, st>>>(S.topb, S.denom, H, Wd);
  k_top_write<<<nblk((long)HW * 48), 256, 0, st>>>(S.topb, S.denom, outc, H, Wd,
                                                   (long)HW * 48);
}

// ---------------------------------------------------------------------------
// kernel_launch
// Input order (jax pytree flatten => dict keys sorted):
//   0..9   assp: a12_b,a12_w,a18_b,a18_w,a6_b,a6_w,ap_b,ap_w,c1_b,c1_w
//   10..16 sa1:  f_b,f_w,g_b,g_w,h_b,h_w,scale          (17..23 sa2, 24..30 sa3)
//   31..40 up:   b1..b5, w1..w5
//   41..53 vgg_bs[0..12],  54..66 vgg_ws[0..12],  67: x (1,3,512,512)
// ---------------------------------------------------------------------------
extern "C" void kernel_launch(void* const* d_in, const int* in_sizes, int n_in,
                              void* d_out, int out_size, void* d_ws, size_t ws_size,
                              hipStream_t stream) {
  (void)in_sizes; (void)n_in; (void)out_size; (void)ws_size;

  const float* x = (const float*)d_in[67];
  const float* vb[13]; const float* vw[13];
  for (int i = 0; i < 13; ++i) {
    vb[i] = (const float*)d_in[41 + i];
    vw[i] = (const float*)d_in[54 + i];
  }
  const float* ub[5]; const float* uw[5];
  for (int i = 0; i < 5; ++i) {
    ub[i] = (const float*)d_in[31 + i];
    uw[i] = (const float*)d_in[36 + i];
  }

  // workspace bump allocator
  char* base = (char*)d_ws;
  size_t off = 0;
  auto alloc = [&](size_t bytes) -> void* {
    void* p = base + off;
    off = (off + bytes + 511) & ~(size_t)511;
    return p;
  };
  float*          actA   = (float*)alloc(64UL * 512 * 512 * 4);   // 67 MB
  float*          actB   = (float*)alloc(64UL * 512 * 512 * 4);   // 67 MB
  float*          f3     = (float*)alloc(256UL * 4096 * 4);
  float*          f4     = (float*)alloc(512UL * 4096 * 4);
  float*          f5     = (float*)alloc(512UL * 4096 * 4);
  unsigned short* wbuf   = (unsigned short*)alloc(512UL * 512 * 9 * 2);
  unsigned short* colb   = (unsigned short*)alloc((size_t)COLCAP * 2); // 96 MB
  float*          sbuf   = (float*)alloc(4096UL * 4096 * 4);      // 67 MB
  unsigned short* betab  = (unsigned short*)alloc(4096UL * 4096 * 2);
  float*          fnf    = (float*)alloc(512UL * 4096 * 4);
  unsigned short* fnb    = (unsigned short*)alloc(512UL * 4096 * 2);
  float*          fo     = (float*)alloc(64UL * 4096 * 4);
  float*          go     = (float*)alloc(64UL * 4096 * 4);
  float*          ho     = (float*)alloc(512UL * 4096 * 4);
  unsigned short* fT     = (unsigned short*)alloc(4096UL * 64 * 2);
  unsigned short* gT     = (unsigned short*)alloc(64UL * 4096 * 2);
  unsigned short* hT     = (unsigned short*)alloc(4096UL * 512 * 2);
  float*          obuf   = (float*)alloc(4096UL * 512 * 4);
  float*          fdd    = (float*)alloc(4096UL * 512 * 4);
  unsigned short* fddb   = (unsigned short*)alloc(4096UL * 512 * 2);
  float*          topb   = (float*)alloc(4096UL * 48 * 4);
  float*          denom  = (float*)alloc(64UL * 48 * 4);
  float*          concat = (float*)alloc(144UL * 4096 * 4);
  float*          asspo  = (float*)alloc(240UL * 4096 * 4);
  float*          avgb   = (float*)alloc(256 * 4);

  // -------------------- VGG backbone --------------------
  conv_layer(x,    3,   512, 512, vw[0],  vb[0],  64,  3, 1, 1, true, actA, wbuf, colb, stream);
  conv_layer(actA, 64,  512, 512, vw[1],  vb[1],  64,  3, 1, 1, true, actB, wbuf, colb, stream);
  k_maxpool2<<<nblk(64L * 256 * 256), 256, 0, stream>>>(actB, actA, 512, 512, 64L * 256 * 256);
  conv_layer(actA, 64,  256, 256, vw[2],  vb[2],  128, 3, 1, 1, true, actB, wbuf, colb, stream);
  conv_layer(actB, 128, 256, 256, vw[3],  vb[3],  128, 3, 1, 1, true, actA, wbuf, colb, stream);
  k_maxpool2<<<nblk(128L * 128 * 128), 256, 0, stream>>>(actA, actB, 256, 256, 128L * 128 * 128);
  conv_layer(actB, 128, 128, 128, vw[4],  vb[4],  256, 3, 1, 1, true, actA, wbuf, colb, stream);
  conv_layer(actA, 256, 128, 128, vw[5],  vb[5],  256, 3, 1, 1, true, actB, wbuf, colb, stream);
  conv_layer(actB, 256, 128, 128, vw[6],  vb[6],  256, 3, 1, 1, true, actA, wbuf, colb, stream);
  k_maxpool2<<<nblk(256L * 64 * 64), 256, 0, stream>>>(actA, f3, 128, 128, 256L * 64 * 64);
  conv_layer(f3,   256, 64, 64,   vw[7],  vb[7],  512, 3, 1, 1, true, actA, wbuf, colb, stream);
  conv_layer(actA, 512, 64, 64,   vw[8],  vb[8],  512, 3, 1, 1, true, actB, wbuf, colb, stream);
  conv_layer(actB, 512, 64, 64,   vw[9],  vb[9],  512, 3, 1, 1, true, f4,   wbuf, colb, stream);
  conv_layer(f4,   512, 64, 64,   vw[10], vb[10], 512, 3, 2, 2, true, actA, wbuf, colb, stream);
  conv_layer(actA, 512, 64, 64,   vw[11], vb[11], 512, 3, 2, 2, true, actB, wbuf, colb, stream);
  conv_layer(actB, 512, 64, 64,   vw[12], vb[12], 512, 3, 2, 2, true, f5,   wbuf, colb, stream);

  // -------------------- SCSA x3 -> concat (144, 64, 64) --------------------
  Scratch S{wbuf, colb, sbuf, betab, fnf, fnb, fo, go, ho,
            fT, gT, hT, obuf, fdd, fddb, topb, denom};
  scsa(f3, 256, d_in, 10, concat,               S, stream);
  scsa(f4, 512, d_in, 17, concat + 48L * 4096,  S, stream);
  scsa(f5, 512, d_in, 24, concat + 96L * 4096,  S, stream);

  // -------------------- ASSP -> (240, 64, 64) --------------------
  conv_layer(concat, 144, 64, 64, (const float*)d_in[9], (const float*)d_in[8],
             48, 1, 0, 1,  true, asspo,               wbuf, colb, stream);
  conv_layer(concat, 144, 64, 64, (const float*)d_in[5], (const float*)d_in[4],
             48, 3, 6, 6,  true, asspo + 48L * 4096,  wbuf, colb, stream);
  conv_layer(concat, 144, 64, 64, (const float*)d_in[1], (const float*)d_in[0],
             48, 3, 12, 12, true, asspo + 96L * 4096, wbuf, colb, stream);
  conv_layer(concat, 144, 64, 64, (const float*)d_in[3], (const float*)d_in[2],
             48, 3, 18, 18, true, asspo + 144L * 4096, wbuf, colb, stream);
  k_channel_mean<<<144, 256, 0, stream>>>(concat, avgb, 4096);
  k_assp_pool<<<48, 256, 0, stream>>>(avgb, (const float*)d_in[7],
                                      (const float*)d_in[6],
                                      asspo + 192L * 4096, 144, 4096);

  // -------------------- Decoder --------------------
  conv_layer(asspo, 240, 64, 64, uw[0], ub[0], 96, 1, 0, 1, true, actA, wbuf, colb, stream);
  k_upsample2<<<nblk(96L * 128 * 128), 256, 0, stream>>>(actA, actB, 64, 64, 96L * 128 * 128);
  conv_layer(actB, 96, 128, 128, uw[1], ub[1], 240, 1, 0, 1, true, actA, wbuf, colb, stream);
  k_upsample2<<<nblk(240L * 256 * 256), 256, 0, stream>>>(actA, actB, 128, 128, 240L * 256 * 256);
  conv_layer(actB, 240, 256, 256, uw[2], ub[2], 240, 3, 1, 1, true, actA, wbuf, colb, stream);
  conv_layer(actA, 240, 256, 256, uw[3], ub[3], 240, 3, 1, 1, true, actB, wbuf, colb, stream);
  conv_layer(actB, 240, 256, 256, uw[4], ub[4], 1,   1, 0, 1, true, actA, wbuf, colb, stream);
  k_upsample2<<<nblk(1L * 512 * 512), 256, 0, stream>>>(actA, (float*)d_out, 256, 256,
                                                        1L * 512 * 512);
}
